// DCBlock_88381837017390
// MI455X (gfx1250) — matrix operand order repaired and checked by reading, not compiled, not run on hardware
//
#include <hip/hip_runtime.h>
#include <hip/hip_bf16.h>

typedef __attribute__((ext_vector_type(16))) _Float16 v16h;
typedef __attribute__((ext_vector_type(8)))  float    v8f;

#define C_CH   256
#define HW     6400
#define H_DIM  80
#define W_DIM  80
#define CP     19
#define KTAPS  49
#define NT     400   // HW/16 N-tiles
#define KT     8     // C/32  K-tiles
#define MT     16    // C/16  M-tiles

// ---------------------------------------------------------------------------
// B-fragment packed index for V_WMMA_F32_16X16X32_F16 (B is 32x16, K x N):
//   lane = 16*(k%32 / 16) + (p%16),  half-slot = (k%32) % 16
//   fragment id = kt*NT + nt ; each fragment = 32 lanes * 16 halves (1 KB)
// ---------------------------------------------------------------------------
__device__ __forceinline__ int b_pack_idx(int c, int p) {
    int kt = c >> 5, kk = c & 31;
    int khalf = kk >> 4, klocal = kk & 15;
    int nt = p >> 4, n = p & 15;
    return (((kt * NT + nt) * 32) + (khalf * 16 + n)) * 16 + klocal;
}

// ---------------------------------------------------------------------------
// Kernel 1: pack a 256x256 fp32 weight (row-major [o][c]) into A-fragment
// layout (16x32 f16 tile, contiguous k order used here):
//   lanes 0-15: M=lane, K=h (half-slot 0..15) ; lanes 16-31: K=16+h
// ---------------------------------------------------------------------------
__global__ __launch_bounds__(256)
void pack_weight_kernel(const float* __restrict__ w, _Float16* __restrict__ apk) {
    int i = blockIdx.x * blockDim.x + threadIdx.x;    // 16*8*32*16 = 65536
    if (i >= MT * KT * 32 * 16) return;
    int h    = i & 15;
    int lane = (i >> 4) & 31;
    int kt   = (i >> 9) & 7;
    int mt   = i >> 12;
    int m     = mt * 16 + (lane & 15);
    int khalf = lane >> 4;
    int v = h >> 1, e = h & 1;
    int kin = 16 * khalf + 2 * v + e;
    int k = kt * 32 + kin;
    apk[i] = (_Float16)w[m * C_CH + k];
}

// ---------------------------------------------------------------------------
// Kernel 2: BatchNorm (inference) + f16 convert + store in B-fragment layout
// ---------------------------------------------------------------------------
__global__ __launch_bounds__(256)
void bn_pack_kernel(const float* __restrict__ x,
                    const float* __restrict__ gamma, const float* __restrict__ beta,
                    const float* __restrict__ mean,  const float* __restrict__ var,
                    _Float16* __restrict__ bpk) {
    int i = blockIdx.x * blockDim.x + threadIdx.x;    // c*HW + p
    if (i >= C_CH * HW) return;
    int c = i / HW, p = i - c * HW;
    float inv = rsqrtf(var[c] + 1e-5f);
    float v = (x[i] - mean[c]) * (inv * gamma[c]) + beta[c];
    bpk[b_pack_idx(c, p)] = (_Float16)v;
}

// ---------------------------------------------------------------------------
// Kernel 3: affinity = softmax_k( exp(-dist_sq_k / (2*relu(sigma)^2 + 1e-8)) )
// one thread per pixel; two passes over its own 49 global slots
// ---------------------------------------------------------------------------
__global__ __launch_bounds__(256)
void affinity_kernel(const float* __restrict__ cp, const float* __restrict__ sigma,
                     float* __restrict__ aff) {
    int p = blockIdx.x * blockDim.x + threadIdx.x;
    if (p >= HW) return;
    int y = p / W_DIM, x = p - y * W_DIM;
    float s = fmaxf(sigma[0], 0.0f);
    float invDenom = 1.0f / (2.0f * s * s + 1e-8f);

    float ctr[CP];
#pragma unroll
    for (int c = 0; c < CP; ++c) ctr[c] = cp[c * HW + p];

    float sum = 0.0f;
    for (int kk = 0; kk < KTAPS; ++kk) {
        int dy = kk / 7 - 3, dx = kk % 7 - 3;
        int yy = y + dy, xx = x + dx;
        bool in = (yy >= 0) & (yy < H_DIM) & (xx >= 0) & (xx < W_DIM);
        int np = yy * W_DIM + xx;
        float d = 0.0f;
#pragma unroll
        for (int c = 0; c < CP; ++c) {
            float v = in ? cp[c * HW + np] : 0.0f;
            float t = v - ctr[c];
            d += t * t;
        }
        float w  = __expf(-d * invDenom);
        float ew = __expf(w);            // softmax of w => exp(w)/sum(exp(w))
        aff[kk * HW + p] = ew;
        sum += ew;
    }
    float inv = 1.0f / sum;
    for (int kk = 0; kk < KTAPS; ++kk)
        aff[kk * HW + p] *= inv;         // same-thread RAW on global: ordered
}

// ---------------------------------------------------------------------------
// Kernel 4: WMMA GEMM  D[256 x 6400] = A[256 x 256] * B[256 x 6400]
// 8 waves / block; wave owns 2 M-tiles x 4 N-tiles; loops 8 K-tiles.
// B fragments are staged into LDS with global_load_async_to_lds_b128
// (ASYNCcnt-tracked, double-buffered); A fragments stream from global/L0.
// mode 0: store f16 messages (c-major).  mode 1: out = addSrc + acc (fp32).
// ---------------------------------------------------------------------------
__global__ __launch_bounds__(256)
void gemm_wmma_kernel(const _Float16* __restrict__ apk,
                      const _Float16* __restrict__ bpk,
                      const float*    __restrict__ addSrc,
                      _Float16*       __restrict__ outH,
                      float*          __restrict__ outF,
                      int mode) {
    __shared__ __align__(32) char smemB[2][4096];   // double-buffered 4 x 1KB frags

    int tid  = threadIdx.x;
    int lane = tid & 31;
    int wave = tid >> 5;
    int ntb  = blockIdx.x * 4;           // 4 N-tiles per block
    int mt0  = wave * 2;

    // async staging: thread t copies 16B chunk (t&63) of fragment (t>>6);
    // LDS dest offset j*1024 + chunk*16 == tid*16 (linear copy)
    int jfrag = tid >> 6;
    int chunk = tid & 63;
    const char* gB = (const char*)bpk;

    auto issue_stage = [&](int kt, int buf) {
        const char* src = gB + (((size_t)(kt * NT + ntb + jfrag)) << 10) + ((size_t)chunk << 4);
        unsigned ldsOff = (unsigned)(uintptr_t)(&smemB[buf][0]) + (unsigned)(tid << 4);
        asm volatile("global_load_async_to_lds_b128 %0, %1, off"
                     :: "v"(ldsOff), "v"((unsigned long long)(uintptr_t)src)
                     : "memory");
    };

    v8f zero = {};
    v8f acc[2][4];
#pragma unroll
    for (int mi = 0; mi < 2; ++mi)
#pragma unroll
        for (int j = 0; j < 4; ++j) acc[mi][j] = zero;

    issue_stage(0, 0);                   // prologue: stages 0 and 1 in flight
    issue_stage(1, 1);
    int buf = 0;

#pragma unroll
    for (int kt = 0; kt < KT; ++kt) {
        // stage kt complete when only the (kt+1) prefetch remains outstanding
        if (kt < KT - 1) asm volatile("s_wait_asynccnt 1" ::: "memory");
        else             asm volatile("s_wait_asynccnt 0" ::: "memory");
        __syncthreads();                 // all waves' stage-kt data visible

        v16h a0 = *(const v16h*)(apk + (size_t)(((mt0 + 0) * KT + kt) * 32 + lane) * 16);
        v16h a1 = *(const v16h*)(apk + (size_t)(((mt0 + 1) * KT + kt) * 32 + lane) * 16);

        const char* sb = &smemB[buf][0];
#pragma unroll
        for (int j = 0; j < 4; ++j) {
            v16h b = *(const v16h*)(sb + j * 1024 + lane * 32);   // ds_load_b128 x2
            acc[0][j] = __builtin_amdgcn_wmma_f32_16x16x32_f16(
                false, a0, false, b, (short)0, acc[0][j], false, false);
            acc[1][j] = __builtin_amdgcn_wmma_f32_16x16x32_f16(
                false, a1, false, b, (short)0, acc[1][j], false, false);
        }

        __syncthreads();                 // buffer fully consumed by all waves
        if (kt + 2 < KT) issue_stage(kt + 2, buf);
        buf ^= 1;
    }

    // D layout: VGPR r, lanes 0-15 -> M=r ; lanes 16-31 -> M=r+8 ; N = lane%16
    int khalf = lane >> 4, nlo = lane & 15;
#pragma unroll
    for (int mi = 0; mi < 2; ++mi) {
#pragma unroll
        for (int j = 0; j < 4; ++j) {
#pragma unroll
            for (int r = 0; r < 8; ++r) {
                int m = (mt0 + mi) * 16 + r + 8 * khalf;
                int n = (ntb + j) * 16 + nlo;
                size_t o = (size_t)m * HW + n;
                if (mode == 0) outH[o] = (_Float16)acc[mi][j][r];
                else           outF[o] = addSrc[o] + acc[mi][j][r];
            }
        }
    }
}

// ---------------------------------------------------------------------------
// Kernel 5: agg[c,p] = sum_k aff[k,p] * msg[c, p + offset(k)]  (zero pad)
// writes result straight into B-fragment layout for GEMM2
// ---------------------------------------------------------------------------
__global__ __launch_bounds__(256)
void aggregate_kernel(const _Float16* __restrict__ msg,
                      const float*    __restrict__ aff,
                      _Float16*       __restrict__ bpk) {
    int i = blockIdx.x * blockDim.x + threadIdx.x;   // c*HW + p
    if (i >= C_CH * HW) return;
    int c = i / HW, p = i - c * HW;
    int y = p / W_DIM, x = p - y * W_DIM;
    const _Float16* mrow = msg + (size_t)c * HW;

    float sum = 0.0f;
    for (int kk = 0; kk < KTAPS; ++kk) {
        int dy = kk / 7 - 3, dx = kk % 7 - 3;
        int yy = y + dy, xx = x + dx;
        bool in = (yy >= 0) & (yy < H_DIM) & (xx >= 0) & (xx < W_DIM);
        float m = in ? (float)mrow[yy * W_DIM + xx] : 0.0f;
        sum += aff[kk * HW + p] * m;
    }
    bpk[b_pack_idx(c, p)] = (_Float16)sum;
}

// ---------------------------------------------------------------------------
extern "C" void kernel_launch(void* const* d_in, const int* in_sizes, int n_in,
                              void* d_out, int out_size, void* d_ws, size_t ws_size,
                              hipStream_t stream) {
    const float* x      = (const float*)d_in[0];
    const float* cp     = (const float*)d_in[1];
    const float* sigma  = (const float*)d_in[2];
    const float* w_feat = (const float*)d_in[3];
    const float* w_fuse = (const float*)d_in[4];
    const float* gamma  = (const float*)d_in[5];
    const float* beta   = (const float*)d_in[6];
    const float* mean   = (const float*)d_in[7];
    const float* var    = (const float*)d_in[8];
    float* out = (float*)d_out;

    char* ws = (char*)d_ws;
    _Float16* apk_feat = (_Float16*)(ws);                      // 131072 B
    _Float16* apk_fuse = (_Float16*)(ws + 131072);             // 131072 B
    _Float16* bpk      = (_Float16*)(ws + 262144);             // 3,276,800 B (xn, then agg)
    float*    aff      = (float*)   (ws + 3538944);            // 1,254,400 B
    _Float16* msg      = (_Float16*)(ws + 4793600);            // 3,276,800 B  (total ~8.07 MB)

    // 1. pack both weights into A-fragment layout (f16)
    pack_weight_kernel<<<256, 256, 0, stream>>>(w_feat, apk_feat);
    pack_weight_kernel<<<256, 256, 0, stream>>>(w_fuse, apk_fuse);

    // 2. BN + pack xn into B-fragment layout
    bn_pack_kernel<<<(C_CH * HW) / 256, 256, 0, stream>>>(x, gamma, beta, mean, var, bpk);

    // 3. affinity weights
    affinity_kernel<<<HW / 256, 256, 0, stream>>>(cp, sigma, aff);

    // 4. messages = w_feat @ xn   (WMMA, f16 in / f32 acc, f16 out)
    gemm_wmma_kernel<<<NT / 4, 256, 0, stream>>>(apk_feat, bpk, nullptr, msg, nullptr, 0);

    // 5. weighted 49-tap aggregation -> B-fragment layout (reuses bpk)
    aggregate_kernel<<<(C_CH * HW) / 256, 256, 0, stream>>>(msg, aff, bpk);

    // 6. out = x + w_fuse @ agg   (WMMA)
    gemm_wmma_kernel<<<NT / 4, 256, 0, stream>>>(apk_fuse, bpk, x, nullptr, out, 1);
}
